// CUDASeasonalityModel_23639499997397
// MI455X (gfx1250) — matrix operand order repaired and checked by reading, not compiled, not run on hardware
//
#include <hip/hip_runtime.h>

// Problem constants (match reference)
#define BB 512
#define TT 4032
#define FF 64
#define NSEAS 10
#define HH 22          // hidden size
#define GG 88          // 4*H gates
#define GSTR 100       // LDS row stride for gates (16 rows x 96 used cols)
#define HSTR 24        // LDS row stride for h (22 cols + 2 zero pad)

typedef __attribute__((ext_vector_type(16))) __bf16        v16bf;
typedef __attribute__((ext_vector_type(8)))  float         v8f;
typedef __attribute__((ext_vector_type(2)))  float         v2f;
typedef __attribute__((ext_vector_type(4)))  unsigned int  u32x4;
typedef __attribute__((ext_vector_type(8)))  int           i32x8;
typedef __attribute__((ext_vector_type(4)))  int           i32x4;

__device__ __forceinline__ float sigmoidf_(float x) {
  return 1.0f / (1.0f + __expf(-x));
}

// One wave (32 lanes) per 16-row batch tile. Runs the full T=4032 recurrence.
// Gates(16x88) = x_t(16x64)@W_ih^T  [bf16 WMMA, f32 acc]
//             +  h(16x22)@W_hh^T    [f32 16x16x4 WMMA]
//             +  (b_ih + b_hh)
__global__ __launch_bounds__(32) void lstm_kernel(
    const float* __restrict__ x, const float* __restrict__ Wih,
    const float* __restrict__ Whh, const float* __restrict__ bih,
    const float* __restrict__ bhh, float* __restrict__ hlast)
{
  __shared__ float xbuf[2][16 * FF];   // double-buffered x tile (TDM dest)
  __shared__ float gates[16 * GSTR];   // gate scratch for cross-lane mixing
  __shared__ float hbuf[16 * HSTR];    // h state (cols 22,23 stay zero)

  const int  lane = threadIdx.x;
  const int  ln15 = lane & 15;
  const bool hi   = lane >= 16;
  const int  b0   = blockIdx.x * 16;

  // ---- one-time: pre-swizzle weight fragments into registers ----
  // B-operand bf16 16x16x32: element e -> K = 32*ks + e + (lane<16 ? 0 : 16)
  v16bf wihF[2][6];
  #pragma unroll
  for (int ks = 0; ks < 2; ++ks) {
    #pragma unroll
    for (int nt = 0; nt < 6; ++nt) {
      const int g = nt * 16 + ln15;
      #pragma unroll
      for (int e = 0; e < 16; ++e) {
        const int f = ks * 32 + e + (hi ? 16 : 0);
        const float w = (g < GG) ? Wih[g * FF + f] : 0.0f;
        wihF[ks][nt][e] = (__bf16)w;
      }
    }
  }
  // B-operand f32 4x16: element e -> K = 4*ks + e + (lane<16 ? 0 : 2)
  v2f whhF[6][6];
  #pragma unroll
  for (int ks = 0; ks < 6; ++ks) {
    #pragma unroll
    for (int nt = 0; nt < 6; ++nt) {
      const int g = nt * 16 + ln15;
      #pragma unroll
      for (int e = 0; e < 2; ++e) {
        const int k = ks * 4 + e + (hi ? 2 : 0);
        whhF[ks][nt][e] = (g < GG && k < HH) ? Whh[g * HH + k] : 0.0f;
      }
    }
  }
  // bias (b_ih + b_hh) per gate column; same value for all 8 C rows of a lane
  float biasv[6];
  #pragma unroll
  for (int nt = 0; nt < 6; ++nt) {
    const int g = nt * 16 + ln15;
    biasv[nt] = (g < GG) ? (bih[g] + bhh[g]) : 0.0f;
  }

  // ---- init state ----
  for (int i = lane; i < 16 * HSTR; i += 32) hbuf[i] = 0.0f;
  float cst[11];
  #pragma unroll
  for (int k = 0; k < 11; ++k) cst[k] = 0.0f;
  __syncthreads();

  // ---- TDM: 2D tile load x[b0..b0+15][t][0..63] -> LDS, data_size=4B ----
  auto issue_tdm = [&](int buf, int t) {
    const unsigned int ldsa =
        (unsigned int)(unsigned long long)(&xbuf[buf][0]);
    const unsigned long long ga = (unsigned long long)(const void*)x +
        ((unsigned long long)b0 * TT * FF + (unsigned long long)t * FF) * 4ull;
    u32x4 g0 = {0u, 0u, 0u, 0u};
    g0[0] = 1u;                                   // count=1, user mode
    g0[1] = ldsa;                                 // lds_addr [63:32]
    g0[2] = (unsigned int)(ga & 0xffffffffull);   // global_addr lo
    g0[3] = (unsigned int)((ga >> 32) & 0x01ffffffull) | 0x80000000u; // type=2
    i32x8 g1 = {0, 0, 0, 0, 0, 0, 0, 0};
    g1[0] = (int)(2u << 16);        // data_size = 4 bytes
    g1[1] = (int)(64u << 16);       // tensor_dim0 = 64   (bits 79:48)
    g1[2] = (int)(16u << 16);       // tensor_dim1 = 16   (bits 111:80)
    g1[3] = (int)(64u << 16);       // tile_dim0   = 64   (bits 127:112)
    g1[4] = 16;                     // tile_dim1   = 16   (bits 143:128)
    g1[5] = TT * FF;                // tensor_dim0_stride (bits 207:160)
    i32x4 z4 = {0, 0, 0, 0};
    i32x8 z8 = {0, 0, 0, 0, 0, 0, 0, 0};
    __builtin_amdgcn_tensor_load_to_lds(g0, g1, z4, z4, z8, 0);
  };

  issue_tdm(0, 0);

  for (int t = 0; t < TT; ++t) {
    const int cur = t & 1;
    __syncthreads();                 // drain prior LDS reads of the dest buf
    if (t + 1 < TT) {
      issue_tdm(cur ^ 1, t + 1);     // prefetch next tile while we compute
      __builtin_amdgcn_s_wait_tensorcnt(1);   // oldest (tile t) complete
    } else {
      __builtin_amdgcn_s_wait_tensorcnt(0);
    }
    __syncthreads();                 // make TDM data visible to ds loads

    // C init = broadcast bias
    v8f acc[6];
    #pragma unroll
    for (int nt = 0; nt < 6; ++nt) {
      #pragma unroll
      for (int r = 0; r < 8; ++r) acc[nt][r] = biasv[nt];
    }

    // x part: bf16 WMMA, K = 64 in 2 steps of 32
    const float* xb = &xbuf[cur][0];
    #pragma unroll
    for (int ks = 0; ks < 2; ++ks) {
      v16bf a;
      #pragma unroll
      for (int e = 0; e < 16; ++e) {
        const int koff = e + (hi ? 16 : 0);
        a[e] = (__bf16)xb[ln15 * FF + ks * 32 + koff];
      }
      #pragma unroll
      for (int nt = 0; nt < 6; ++nt)
        acc[nt] = __builtin_amdgcn_wmma_f32_16x16x32_bf16(
            false, a, false, wihF[ks][nt], (short)0, acc[nt], false, false);
    }

    // h part: f32 16x16x4 WMMA, K = 22 padded to 24 in 6 steps of 4
    #pragma unroll
    for (int ks = 0; ks < 6; ++ks) {
      v2f ah;
      #pragma unroll
      for (int e = 0; e < 2; ++e) {
        const int k = ks * 4 + e + (hi ? 2 : 0);  // <= 23 < HSTR, pad is 0
        ah[e] = hbuf[ln15 * HSTR + k];
      }
      #pragma unroll
      for (int nt = 0; nt < 6; ++nt)
        acc[nt] = __builtin_amdgcn_wmma_f32_16x16x4_f32(
            false, ah, false, whhF[ks][nt], (short)0, acc[nt], false, false);
    }

    // spill gates to LDS for cross-lane i/f/g/o mixing
    #pragma unroll
    for (int nt = 0; nt < 6; ++nt) {
      #pragma unroll
      for (int r = 0; r < 8; ++r) {
        const int m = r + (hi ? 8 : 0);
        gates[m * GSTR + nt * 16 + ln15] = acc[nt][r];
      }
    }
    __syncthreads();

    // elementwise LSTM cell: 16*22 = 352 items = 11 per lane
    #pragma unroll
    for (int k = 0; k < 11; ++k) {
      const int idx = lane * 11 + k;
      const int m = idx / 22, j = idx % 22;
      const float* gr = &gates[m * GSTR];
      const float ig = sigmoidf_(gr[j]);
      const float fg = sigmoidf_(gr[22 + j]);
      const float gv = tanhf(gr[44 + j]);
      const float og = sigmoidf_(gr[66 + j]);
      const float c = fg * cst[k] + ig * gv;
      cst[k] = c;
      hbuf[m * HSTR + j] = og * tanhf(c);
    }
  }

  __syncthreads();
  #pragma unroll
  for (int k = 0; k < 11; ++k) {
    const int idx = lane * 11 + k;
    const int m = idx / 22, j = idx % 22;
    hlast[(b0 + m) * HH + j] = hbuf[m * HSTR + j];
  }
}

// out[b][t] = sum_n a[n-1]*cos(2pi*n*Ps*(t+1)) + a[10+n-1]*sin(...) + bias
__global__ __launch_bounds__(256) void seasonal_kernel(
    const float* __restrict__ hlast, float* __restrict__ out)
{
  const int b = blockIdx.x;
  const float* hl = hlast + b * HH;
  float a[2 * NSEAS];
  #pragma unroll
  for (int n = 0; n < 2 * NSEAS; ++n) a[n] = hl[n];
  const float Ps   = hl[NSEAS];
  const float bias = hl[2 * NSEAS + 1];
  const float w0   = 6.283185307179586f * Ps;

  for (int t = threadIdx.x; t < TT; t += blockDim.x) {
    const float tv = (float)(t + 1);
    float accv = bias;
    #pragma unroll
    for (int n = 1; n <= NSEAS; ++n) {
      float s, c;
      __sincosf(w0 * (float)n * tv, &s, &c);
      accv += a[n - 1] * c + a[NSEAS + n - 1] * s;
    }
    out[b * TT + t] = accv;
  }
}

extern "C" void kernel_launch(void* const* d_in, const int* in_sizes, int n_in,
                              void* d_out, int out_size, void* d_ws, size_t ws_size,
                              hipStream_t stream) {
  const float* x   = (const float*)d_in[0];
  // d_in[1] = xTimestampSizes (int64), all == T -> unused
  const float* Wih = (const float*)d_in[2];
  const float* Whh = (const float*)d_in[3];
  const float* bih = (const float*)d_in[4];
  const float* bhh = (const float*)d_in[5];

  float* hlast = (float*)d_ws;   // 512*22 floats of scratch

  lstm_kernel<<<BB / 16, 32, 0, stream>>>(x, Wih, Whh, bih, bhh, hlast);
  seasonal_kernel<<<BB, 256, 0, stream>>>(hlast, (float*)d_out);
}
